// SquaredTaylorAttention_25984552141258
// MI455X (gfx1250) — matrix operand order repaired and checked by reading, not compiled, not run on hardware
//
#include <hip/hip_runtime.h>

typedef __attribute__((ext_vector_type(16))) _Float16 v16h;
typedef __attribute__((ext_vector_type(8)))  _Float16 v8h;
typedef __attribute__((ext_vector_type(8)))  float    v8f;
typedef __attribute__((ext_vector_type(4)))  int      v4i;

#define NQ      2048     // sequence length
#define DH      64       // head dim
#define BH_TOT  24       // B*H
#define BM      128      // query rows per block (8 waves x 16)
#define BN      64       // key tile
#define NWAVES  8
#define LSTRIDE 72       // LDS row stride in halfs: 144B rows keep b128 segs 16B-aligned
#define NTILES  (NQ / BN)

// ---------------- CDNA5 async global->LDS helpers ----------------------------
#if __has_builtin(__builtin_amdgcn_global_load_async_to_lds_b128)
#define HAVE_ASYNC_LDS 1
#else
#define HAVE_ASYNC_LDS 0
#endif

__device__ __forceinline__ void cp16_async(const _Float16* g, _Float16* l) {
#if HAVE_ASYNC_LDS
  __builtin_amdgcn_global_load_async_to_lds_b128(
      (__attribute__((address_space(1))) v4i*)g,
      (__attribute__((address_space(3))) v4i*)l, 0, 0);
#else
  *(v8h*)l = *(const v8h*)g;   // synchronous fallback
#endif
}

#if __has_builtin(__builtin_amdgcn_s_wait_asynccnt)
#define WAIT_ASYNC(n) __builtin_amdgcn_s_wait_asynccnt(n)
#elif HAVE_ASYNC_LDS
#define WAIT_ASYNC(n) asm volatile("s_wait_asynccnt %0" ::"n"(n) : "memory")
#else
#define WAIT_ASYNC(n) do {} while (0)
#endif

// ---------------- WMMA fragment builders -------------------------------------
// B-frag (32x16 f16): lane l holds col n=l%16; half h -> k = 16*(l/16) + h.
__device__ __forceinline__ v16h wmma_b_from_lds(const _Float16* p) {
  v8h lo = *(const v8h*)(p);
  v8h hi = *(const v8h*)(p + 8);
  v16h b;
#pragma unroll
  for (int i = 0; i < 8; ++i) { b[i] = lo[i]; b[i + 8] = hi[i]; }
  return b;
}

// A-frag (16x32 f16): lane l holds row m=l%16; with g=l/16,
// halfs a[0..7] -> k = 32f + 16g + i ; a[8..15] -> k = 32f + 16g + 8 + i.
__device__ __forceinline__ v16h wmma_a_from_h(const _Float16* row, int g, int f) {
  v8h lo = *(const v8h*)(row + 32 * f + 16 * g);
  v8h hi = *(const v8h*)(row + 32 * f + 16 * g + 8);
  v16h a;
#pragma unroll
  for (int i = 0; i < 8; ++i) { a[i] = lo[i]; a[i + 8] = hi[i]; }
  return a;
}

__device__ __forceinline__ v16h wmma_a_from_global_f32(const float* row, int g, int f) {
  float4 x0 = *(const float4*)(row + 32 * f + 16 * g);
  float4 x1 = *(const float4*)(row + 32 * f + 16 * g + 4);
  float4 x2 = *(const float4*)(row + 32 * f + 16 * g + 8);
  float4 x3 = *(const float4*)(row + 32 * f + 16 * g + 12);
  v16h a;
  a[0]  = (_Float16)x0.x; a[1]  = (_Float16)x0.y; a[2]  = (_Float16)x0.z; a[3]  = (_Float16)x0.w;
  a[4]  = (_Float16)x1.x; a[5]  = (_Float16)x1.y; a[6]  = (_Float16)x1.z; a[7]  = (_Float16)x1.w;
  a[8]  = (_Float16)x2.x; a[9]  = (_Float16)x2.y; a[10] = (_Float16)x2.z; a[11] = (_Float16)x2.w;
  a[12] = (_Float16)x3.x; a[13] = (_Float16)x3.y; a[14] = (_Float16)x3.z; a[15] = (_Float16)x3.w;
  return a;
}

__device__ __forceinline__ void cvt16(const float4& a, const float4& b,
                                      const float4& c, const float4& d,
                                      v8h& lo, v8h& hi) {
  lo[0] = (_Float16)a.x; lo[1] = (_Float16)a.y; lo[2] = (_Float16)a.z; lo[3] = (_Float16)a.w;
  lo[4] = (_Float16)b.x; lo[5] = (_Float16)b.y; lo[6] = (_Float16)b.z; lo[7] = (_Float16)b.w;
  hi[0] = (_Float16)c.x; hi[1] = (_Float16)c.y; hi[2] = (_Float16)c.z; hi[3] = (_Float16)c.w;
  hi[4] = (_Float16)d.x; hi[5] = (_Float16)d.y; hi[6] = (_Float16)d.z; hi[7] = (_Float16)d.w;
}

// =============================================================================
// Pre-pass 1: f32 -> f16 in-layout convert (used for Q and K spans)
// =============================================================================
__global__ __launch_bounds__(256)
void cvt_f16_kernel(const float* __restrict__ src, _Float16* __restrict__ dst) {
  size_t i = ((size_t)blockIdx.x * 256 + threadIdx.x) * 8;
  float4 a = *(const float4*)(src + i);
  float4 b = *(const float4*)(src + i + 4);
  v8h h;
  h[0] = (_Float16)a.x; h[1] = (_Float16)a.y; h[2] = (_Float16)a.z; h[3] = (_Float16)a.w;
  h[4] = (_Float16)b.x; h[5] = (_Float16)b.y; h[6] = (_Float16)b.z; h[7] = (_Float16)b.w;
  *(v8h*)(dst + i) = h;
}

// =============================================================================
// Pre-pass 2: V f32 [bh][n][d] -> f16 transposed Vt [bh][d][n]
// =============================================================================
__global__ __launch_bounds__(256)
void vtrans_f16_kernel(const float* __restrict__ V, _Float16* __restrict__ Vt) {
  __shared__ _Float16 T[DH * LSTRIDE];   // [d][key-in-tile]
  const int bh  = blockIdx.y;
  const int n0  = blockIdx.x * 64;
  const int tid = threadIdx.x;
  const int row = tid >> 2;              // 0..63
  const int seg = tid & 3;               // 0..3 (16 floats each)

  const float* vg = V + ((size_t)bh * NQ + n0 + row) * DH + seg * 16;
  float4 v0 = ((const float4*)vg)[0], v1 = ((const float4*)vg)[1];
  float4 v2 = ((const float4*)vg)[2], v3 = ((const float4*)vg)[3];
  const float vv[16] = {v0.x, v0.y, v0.z, v0.w, v1.x, v1.y, v1.z, v1.w,
                        v2.x, v2.y, v2.z, v2.w, v3.x, v3.y, v3.z, v3.w};
#pragma unroll
  for (int i = 0; i < 16; ++i)
    T[(seg * 16 + i) * LSTRIDE + row] = (_Float16)vv[i];
  __syncthreads();
  _Float16* orow = Vt + ((size_t)bh * DH + row) * NQ + n0 + seg * 16;
  *(v8h*)orow       = *(const v8h*)&T[row * LSTRIDE + seg * 16];
  *(v8h*)(orow + 8) = *(const v8h*)&T[row * LSTRIDE + seg * 16 + 8];
}

// =============================================================================
// Main kernel (fast path): f16 inputs, async double-buffered K/V tiles
// =============================================================================
__global__ __launch_bounds__(256)
void taylor_attn_f16(const _Float16* __restrict__ Qh,
                     const _Float16* __restrict__ Kh,
                     const _Float16* __restrict__ Vtg,   // [bh][d][n]
                     float* __restrict__ out) {
  __shared__ _Float16 Kbuf[2][BN * LSTRIDE];            // [key][d]
  __shared__ _Float16 Vbuf[2][DH * LSTRIDE];            // [d][key]
  __shared__ _Float16 Plds[NWAVES * 16 * LSTRIDE];      // per-wave [m][key]

  const int bh   = blockIdx.y;
  const int tid  = threadIdx.x;
  const int lane = tid & 31;
  const int wave = tid >> 5;
  const int ln16 = lane & 15;
  const int g    = lane >> 4;

  const _Float16* Qhead  = Qh  + (size_t)bh * NQ * DH;
  const _Float16* Khead  = Kh  + (size_t)bh * NQ * DH;
  const _Float16* Vthead = Vtg + (size_t)bh * DH * NQ;

  const int m0 = blockIdx.x * BM + wave * 16;

  // Resident Q A-fragments
  const _Float16* qrow = Qhead + (size_t)(m0 + ln16) * DH;
  const v16h aQ0 = wmma_a_from_h(qrow, g, 0);
  const v16h aQ1 = wmma_a_from_h(qrow, g, 1);

  v8f o[4];
#pragma unroll
  for (int j = 0; j < 4; ++j) o[j] = (v8f){};
  float dsum[8] = {0.f, 0.f, 0.f, 0.f, 0.f, 0.f, 0.f, 0.f};
  _Float16* Prow_w = &Plds[(wave * 16) * LSTRIDE];

  // 4 async b128 issues per thread per tile (K x2, V x2), in-order completion
  auto stage = [&](int nt, int buf) {
#pragma unroll
    for (int rep = 0; rep < 2; ++rep) {
      const int u = tid + rep * 256;     // 0..511 : 16B unit
      const int r = u >> 3;              // row 0..63
      const int s = u & 7;               // 8-half segment
      cp16_async(Khead + (size_t)(nt * BN + r) * DH + s * 8,
                 &Kbuf[buf][r * LSTRIDE + s * 8]);
      cp16_async(Vthead + (size_t)r * NQ + nt * BN + s * 8,
                 &Vbuf[buf][r * LSTRIDE + s * 8]);
    }
  };

  stage(0, 0);
  stage(1, 1);
  WAIT_ASYNC(4);          // tile 0's 4 copies (oldest) have landed
  __syncthreads();

  for (int nt = 0; nt < NTILES; ++nt) {
    const int cur = nt & 1;
    const _Float16* kb = Kbuf[cur];
    const _Float16* vb = Vbuf[cur];

    // ---- S = Q K^T : 16 rows x 64 keys, f32 accum --------------------------
    v8f s[4];
#pragma unroll
    for (int c = 0; c < 4; ++c) {
      const _Float16* krow = kb + (c * 16 + ln16) * LSTRIDE;
      v16h blo = wmma_b_from_lds(krow + 16 * g);        // d  0..31
      v16h bhi = wmma_b_from_lds(krow + 32 + 16 * g);   // d 32..63
      v8f acc = (v8f){};
      acc = __builtin_amdgcn_wmma_f32_16x16x32_f16(false, aQ0, false, blo,
                                                   (short)0, acc, false, false);
      acc = __builtin_amdgcn_wmma_f32_16x16x32_f16(false, aQ1, false, bhi,
                                                   (short)0, acc, false, false);
      s[c] = acc;
    }

    // ---- P = 0.5 s^2 + s + 1 (>0) ; denom partials ; stage P (f16) ---------
#pragma unroll
    for (int c = 0; c < 4; ++c) {
#pragma unroll
      for (int r = 0; r < 8; ++r) {
        float x = s[c][r];
        float p = __builtin_fmaf(0.5f * x, x, x) + 1.0f;
        dsum[r] += p;
        Prow_w[(r + 8 * g) * LSTRIDE + c * 16 + ln16] = (_Float16)p;
      }
    }
    // same-wave LDS store->load stays in order (DScnt in-order per ISA)

    // ---- O += P @ V --------------------------------------------------------
    const _Float16* prow = &Plds[(wave * 16 + ln16) * LSTRIDE];
    v16h pA0 = wmma_a_from_h(prow, g, 0);   // keys  0..31
    v16h pA1 = wmma_a_from_h(prow, g, 1);   // keys 32..63
#pragma unroll
    for (int j = 0; j < 4; ++j) {
      const _Float16* vrow = vb + (j * 16 + ln16) * LSTRIDE;
      v16h blo = wmma_b_from_lds(vrow + 16 * g);
      v16h bhi = wmma_b_from_lds(vrow + 32 + 16 * g);
      o[j] = __builtin_amdgcn_wmma_f32_16x16x32_f16(false, pA0, false, blo,
                                                    (short)0, o[j], false, false);
      o[j] = __builtin_amdgcn_wmma_f32_16x16x32_f16(false, pA1, false, bhi,
                                                    (short)0, o[j], false, false);
    }

    __syncthreads();                        // everyone done reading buf `cur`
    if (nt + 2 < NTILES) {
      stage(nt + 2, cur);                   // refill the just-freed buffer
      WAIT_ASYNC(4);                        // tile nt+1 (older 4) complete
    } else {
      WAIT_ASYNC(0);                        // drain tail
    }
    __syncthreads();                        // tile nt+1 visible to all waves
  }

  // ---- finalize: reduce denominators over 16-lane halves, divide, store ----
  float inv[8];
#pragma unroll
  for (int r = 0; r < 8; ++r) {
    float d = dsum[r];
    d += __shfl_xor(d, 1);
    d += __shfl_xor(d, 2);
    d += __shfl_xor(d, 4);
    d += __shfl_xor(d, 8);
    inv[r] = 1.0f / d;
  }
  float* orow = out + (size_t)bh * NQ * DH;
#pragma unroll
  for (int j = 0; j < 4; ++j) {
#pragma unroll
    for (int r = 0; r < 8; ++r) {
      orow[(size_t)(m0 + r + 8 * g) * DH + j * 16 + ln16] = o[j][r] * inv[r];
    }
  }
}

// =============================================================================
// Fallback (ws too small): round-1 single-pass kernel on f32 inputs
// =============================================================================
__global__ __launch_bounds__(256)
void taylor_attn_f32(const float* __restrict__ QKV, float* __restrict__ out) {
  __shared__ _Float16 Klds[BN * LSTRIDE];
  __shared__ _Float16 Vt[DH * LSTRIDE];
  __shared__ _Float16 Plds[NWAVES * 16 * LSTRIDE];

  const int bh   = blockIdx.y;
  const int tid  = threadIdx.x;
  const int lane = tid & 31;
  const int wave = tid >> 5;
  const int ln16 = lane & 15;
  const int g    = lane >> 4;

  const size_t head = (size_t)bh * NQ * DH;
  const float* Q = QKV + head;
  const float* K = QKV + (size_t)BH_TOT * NQ * DH + head;
  const float* V = QKV + 2 * (size_t)BH_TOT * NQ * DH + head;

  const int m0 = blockIdx.x * BM + wave * 16;
  const float* qrow = Q + (size_t)(m0 + ln16) * DH;
  const v16h aQ0 = wmma_a_from_global_f32(qrow, g, 0);
  const v16h aQ1 = wmma_a_from_global_f32(qrow, g, 1);

  v8f o[4];
#pragma unroll
  for (int j = 0; j < 4; ++j) o[j] = (v8f){};
  float dsum[8] = {0.f, 0.f, 0.f, 0.f, 0.f, 0.f, 0.f, 0.f};

  const int row = tid >> 2;
  const int seg = tid & 3;
  _Float16* Prow_w = &Plds[(wave * 16) * LSTRIDE];

  for (int nt = 0; nt < NTILES; ++nt) {
    const int n0 = nt * BN;
    {
      const float* kg = K + (size_t)(n0 + row) * DH + seg * 16;
      const float* vg = V + (size_t)(n0 + row) * DH + seg * 16;
      float4 k0 = ((const float4*)kg)[0], k1 = ((const float4*)kg)[1];
      float4 k2 = ((const float4*)kg)[2], k3 = ((const float4*)kg)[3];
      v8h hlo, hhi;
      cvt16(k0, k1, k2, k3, hlo, hhi);
      *(v8h*)&Klds[row * LSTRIDE + seg * 16]     = hlo;
      *(v8h*)&Klds[row * LSTRIDE + seg * 16 + 8] = hhi;

      float4 v0 = ((const float4*)vg)[0], v1 = ((const float4*)vg)[1];
      float4 v2 = ((const float4*)vg)[2], v3 = ((const float4*)vg)[3];
      const float vv[16] = {v0.x, v0.y, v0.z, v0.w, v1.x, v1.y, v1.z, v1.w,
                            v2.x, v2.y, v2.z, v2.w, v3.x, v3.y, v3.z, v3.w};
      const int dbase = seg * 16;
#pragma unroll
      for (int i = 0; i < 16; ++i)
        Vt[(dbase + i) * LSTRIDE + row] = (_Float16)vv[i];
    }
    __syncthreads();

    v8f s[4];
#pragma unroll
    for (int c = 0; c < 4; ++c) {
      const _Float16* krow = &Klds[(c * 16 + ln16) * LSTRIDE];
      v16h blo = wmma_b_from_lds(krow + 16 * g);
      v16h bhi = wmma_b_from_lds(krow + 32 + 16 * g);
      v8f acc = (v8f){};
      acc = __builtin_amdgcn_wmma_f32_16x16x32_f16(false, aQ0, false, blo,
                                                   (short)0, acc, false, false);
      acc = __builtin_amdgcn_wmma_f32_16x16x32_f16(false, aQ1, false, bhi,
                                                   (short)0, acc, false, false);
      s[c] = acc;
    }

#pragma unroll
    for (int c = 0; c < 4; ++c) {
#pragma unroll
      for (int r = 0; r < 8; ++r) {
        float x = s[c][r];
        float p = __builtin_fmaf(0.5f * x, x, x) + 1.0f;
        dsum[r] += p;
        Prow_w[(r + 8 * g) * LSTRIDE + c * 16 + ln16] = (_Float16)p;
      }
    }

    const _Float16* prow = &Plds[(wave * 16 + ln16) * LSTRIDE];
    v16h pA0 = wmma_a_from_h(prow, g, 0);
    v16h pA1 = wmma_a_from_h(prow, g, 1);
#pragma unroll
    for (int j = 0; j < 4; ++j) {
      const _Float16* vrow = &Vt[(j * 16 + ln16) * LSTRIDE];
      v16h blo = wmma_b_from_lds(vrow + 16 * g);
      v16h bhi = wmma_b_from_lds(vrow + 32 + 16 * g);
      o[j] = __builtin_amdgcn_wmma_f32_16x16x32_f16(false, pA0, false, blo,
                                                    (short)0, o[j], false, false);
      o[j] = __builtin_amdgcn_wmma_f32_16x16x32_f16(false, pA1, false, bhi,
                                                    (short)0, o[j], false, false);
    }
    __syncthreads();
  }

  float inv[8];
#pragma unroll
  for (int r = 0; r < 8; ++r) {
    float d = dsum[r];
    d += __shfl_xor(d, 1);
    d += __shfl_xor(d, 2);
    d += __shfl_xor(d, 4);
    d += __shfl_xor(d, 8);
    inv[r] = 1.0f / d;
  }
  float* orow = out + head;
#pragma unroll
  for (int j = 0; j < 4; ++j) {
#pragma unroll
    for (int r = 0; r < 8; ++r) {
      orow[(size_t)(m0 + r + 8 * g) * DH + j * 16 + ln16] = o[j][r] * inv[r];
    }
  }
}

// ---- harness entry ----------------------------------------------------------
extern "C" void kernel_launch(void* const* d_in, const int* in_sizes, int n_in,
                              void* d_out, int out_size, void* d_ws, size_t ws_size,
                              hipStream_t stream) {
  (void)in_sizes; (void)n_in; (void)out_size;
  const float* qkv = (const float*)d_in[0];
  float* out = (float*)d_out;

  const size_t SZ = (size_t)BH_TOT * NQ * DH;       // elements per tensor
  const size_t need = 3 * SZ * sizeof(_Float16);    // ~18 MB f16 scratch

  dim3 grid(NQ / BM, BH_TOT);                       // (16, 24), 8 wave32 each

  if (ws_size >= need) {
    _Float16* wsh = (_Float16*)d_ws;
    _Float16* Qh  = wsh;                            // [bh][n][d]
    _Float16* Kh  = wsh + SZ;                       // [bh][n][d]
    _Float16* Vtg = wsh + 2 * SZ;                   // [bh][d][n]
    // pre-pass: convert Q and K spans in-layout (2*SZ elements, 8 per thread)
    cvt_f16_kernel<<<(2 * SZ) / (256 * 8), 256, 0, stream>>>(qkv, Qh);
    // pre-pass: convert + transpose V per head
    vtrans_f16_kernel<<<dim3(NQ / 64, BH_TOT), 256, 0, stream>>>(qkv + 2 * SZ, Vtg);
    taylor_attn_f16<<<grid, 256, 0, stream>>>(Qh, Kh, Vtg, out);
  } else {
    taylor_attn_f32<<<grid, 256, 0, stream>>>(qkv, out);
  }
}
